// VectorQuantizer_24867860644464
// MI455X (gfx1250) — matrix operand order repaired and checked by reading, not compiled, not run on hardware
//
#include <hip/hip_runtime.h>

typedef __attribute__((ext_vector_type(16))) __bf16 v16bf;
typedef __attribute__((ext_vector_type(8)))  float  v8f;

#define NUM_E    1024
#define C_DIM    256
#define DHW      16384          // 16*32*32
#define N_ROWS   65536          // 4*DHW
#define ZQ_ELEMS 16777216       // 4*256*DHW
#define TILE_N   128            // rows per block (8 waves x 16 rows)
#define SHP      272            // padded LDS row stride (bf16 elems), 544 B
#define BSLOT    17408          // bytes per B double-buffer slot (hi 8704 + lo 8704)
#define AHALF    17408          // bytes of A-staging hi (32 rows * 544 B)

static __device__ __forceinline__ unsigned short f2bf(float f) {
  unsigned u = __float_as_uint(f);
  unsigned r = u + 0x7FFFu + ((u >> 16) & 1u);   // round-to-nearest-even
  return (unsigned short)(r >> 16);
}
static __device__ __forceinline__ float bf2f(unsigned short h) {
  return __uint_as_float(((unsigned)h) << 16);
}
struct U4x2 { uint4 a, b; };
static __device__ __forceinline__ v16bf mk16(uint4 a, uint4 b) {
  U4x2 t{a, b};
  return __builtin_bit_cast(v16bf, t);
}
// ISA: flat address of LDS = {shared aperture, offset[31:0]} -> low 32 bits are
// the wave-relative LDS offset the async-LDS ops expect.
static __device__ __forceinline__ unsigned lds_off(const void* p) {
  return (unsigned)(uintptr_t)p;
}

// ---------------- prep: emb -> bf16 hi/lo, e_sq, zero accumulators ----------
__global__ void vq_prep(const float* __restrict__ emb,
                        unsigned short* __restrict__ ehi,
                        unsigned short* __restrict__ elo,
                        float* __restrict__ esq,
                        float* __restrict__ counts,
                        float* __restrict__ gsum) {
  const int k = blockIdx.x;     // 0..1023
  const int c = threadIdx.x;    // 0..255
  float v = emb[k * C_DIM + c];
  unsigned short h = f2bf(v);
  ehi[k * C_DIM + c] = h;
  elo[k * C_DIM + c] = f2bf(v - bf2f(h));
  float s = v * v;
  for (int off = 16; off; off >>= 1) s += __shfl_xor(s, off, 32);
  __shared__ float ws[8];
  if ((threadIdx.x & 31) == 0) ws[threadIdx.x >> 5] = s;
  __syncthreads();
  if (threadIdx.x == 0) {
    float t = 0.f;
    #pragma unroll
    for (int i = 0; i < 8; ++i) t += ws[i];
    esq[k] = t;
    counts[k] = 0.f;
    if (k == 0) *gsum = 0.f;
  }
}

// ---------------- main: WMMA distances + argmin + gather + loss -------------
__global__ void __launch_bounds__(256) vq_main(
    const float* __restrict__ z,
    const float* __restrict__ emb,
    const unsigned short* __restrict__ ehi,
    const unsigned short* __restrict__ elo,
    const float* __restrict__ esq,
    float* __restrict__ out,        // z_q_st (same layout as z)
    float* __restrict__ idx_out,    // encoding indices as float
    float* __restrict__ counts,
    float* __restrict__ gsum) {
  // One buffer, time-shared: phase 1 = A staging (32 rows hi+lo = 34816 B),
  // phase 2 = B double buffer (2 slots x 17408 B).
  __shared__ __align__(32) unsigned char smem[2 * BSLOT];
  __shared__ unsigned long long sbest[TILE_N];

  const int tid  = threadIdx.x;
  const int n0   = blockIdx.x * TILE_N;
  const long long base =
      (long long)(n0 >> 14) * ((long long)C_DIM * DHW) + (n0 & (DHW - 1));

  const int w    = tid >> 5;
  const int lane = tid & 31;
  const int half = lane >> 4;
  const int m    = lane & 15;

  if (tid < TILE_N) sbest[tid] = ~0ull;

  // ---- phase 1: stage z in 4 groups of 32 rows, each wave grabs its A frags
  unsigned short* ahi = (unsigned short*)smem;
  unsigned short* alo = (unsigned short*)(smem + AHALF);
  v16bf afh[8], afl[8];
  {
    const int nl = tid & 31;
    const int cb = tid >> 5;
    for (int g = 0; g < 4; ++g) {
      __syncthreads();            // previous readers done before overwrite
      for (int c = cb; c < C_DIM; c += 8) {
        float v = z[base + (long long)c * DHW + (g * 32 + nl)];
        unsigned short h = f2bf(v);
        ahi[nl * SHP + c] = h;
        alo[nl * SHP + c] = f2bf(v - bf2f(h));
      }
      __syncthreads();
      if ((w >> 1) == g) {
        const int rowA = (16 * (w & 1) + m) * SHP;
        #pragma unroll
        for (int d = 0; d < 8; ++d) {
          const int cA = 32 * d + 16 * half;  // A-layout: K contiguous 16 per lane half
          afh[d] = mk16(*(const uint4*)&ahi[rowA + cA],
                        *(const uint4*)&ahi[rowA + cA + 8]);
          afl[d] = mk16(*(const uint4*)&alo[rowA + cA],
                        *(const uint4*)&alo[rowA + cA + 8]);
        }
      }
    }
    __syncthreads();              // A preloads done; smem now free for B
  }

  // ---- phase 2: sweep all 64 k-tiles; B slab double-buffered in LDS via
  //               async global->LDS copies (ASYNCcnt) overlapped with WMMA.
  const unsigned smem_base = lds_off(smem);
  const unsigned long long ghi = (unsigned long long)(uintptr_t)ehi;
  const unsigned long long glo = (unsigned long long)(uintptr_t)elo;
  // per-thread chunk: row rr (0..15), 32 contiguous bytes at column chunk cc
  const int rr = tid >> 4;
  const int cc = (tid & 15) * 16;                 // ushort offset in row
  const unsigned go_s   = (unsigned)((rr * 256 + cc) * 2);  // + k0*512 per tile
  const unsigned lds_s  = (unsigned)(rr * (SHP * 2) + cc * 2);

  #define STAGE_B(kt, slot)                                                   \
    do {                                                                      \
      unsigned go = go_s + (unsigned)(kt) * (16u * 512u);                     \
      unsigned lh = smem_base + (unsigned)(slot) * BSLOT + lds_s;             \
      unsigned ll = lh + 8704u;                                               \
      asm volatile(                                                           \
          "global_load_async_to_lds_b128 %0, %2, %4\n\t"                      \
          "global_load_async_to_lds_b128 %1, %3, %4"                          \
          :: "v"(lh), "v"(lh + 16u), "v"(go), "v"(go + 16u), "s"(ghi)         \
          : "memory");                                                        \
      asm volatile(                                                           \
          "global_load_async_to_lds_b128 %0, %2, %4\n\t"                      \
          "global_load_async_to_lds_b128 %1, %3, %4"                          \
          :: "v"(ll), "v"(ll + 16u), "v"(go), "v"(go + 16u), "s"(glo)         \
          : "memory");                                                        \
    } while (0)

  unsigned long long best[8];
  #pragma unroll
  for (int j = 0; j < 8; ++j) best[j] = ~0ull;

  STAGE_B(0, 0);
  asm volatile("s_wait_asynccnt 0x0" ::: "memory");
  __syncthreads();

  for (int kt = 0; kt < 64; ++kt) {
    if (kt < 63) STAGE_B(kt + 1, (kt + 1) & 1);

    const int kcol = kt * 16 + m;
    const float es = esq[kcol];
    const unsigned short* bh =
        (const unsigned short*)(smem + (kt & 1) * BSLOT) + m * SHP + 16 * half;
    const unsigned short* bl = bh + (8704 / 2);

    v8f a0 = {}, a1 = {}, a2 = {};
    #pragma unroll
    for (int d = 0; d < 8; ++d) {
      v16bf vbh = mk16(*(const uint4*)(bh + 32 * d),
                       *(const uint4*)(bh + 32 * d + 8));
      v16bf vbl = mk16(*(const uint4*)(bl + 32 * d),
                       *(const uint4*)(bl + 32 * d + 8));
      a0 = __builtin_amdgcn_wmma_f32_16x16x32_bf16(false, afh[d], false, vbh,
                                                   (short)0, a0, false, false);
      a1 = __builtin_amdgcn_wmma_f32_16x16x32_bf16(false, afh[d], false, vbl,
                                                   (short)0, a1, false, false);
      a2 = __builtin_amdgcn_wmma_f32_16x16x32_bf16(false, afl[d], false, vbh,
                                                   (short)0, a2, false, false);
    }
    #pragma unroll
    for (int j = 0; j < 8; ++j) {
      float dot  = a0[j] + a1[j] + a2[j];
      float dist = es - 2.0f * dot;           // x_sq dropped: row-constant
      unsigned ub  = __float_as_uint(dist);
      unsigned key = (ub & 0x80000000u) ? ~ub : (ub | 0x80000000u);
      unsigned long long pk = ((unsigned long long)key << 32) | (unsigned)kcol;
      best[j] = pk < best[j] ? pk : best[j];
    }

    asm volatile("s_wait_asynccnt 0x0" ::: "memory");
    __syncthreads();                          // slot (kt+1)&1 fully staged
  }
  #pragma unroll
  for (int j = 0; j < 8; ++j)
    atomicMin(&sbest[16 * w + 8 * half + j], best[j]);
  __syncthreads();

  // ---- stage 3: gather codebook rows, write z_q_st, loss + histogram
  {
    const int nl = tid & 127;
    const int cb = tid >> 7;                  // 0..1
    const unsigned idx = (unsigned)(sbest[nl] & 0xFFFFFFFFull);
    float lsum = 0.f;
    for (int c = cb; c < C_DIM; c += 2) {
      float e = emb[idx * C_DIM + c];
      long long off = base + (long long)c * DHW + nl;
      float dv = e - z[off];
      out[off] = e;                           // z_q_st == z_q numerically
      lsum += dv * dv;
    }
    for (int o = 16; o; o >>= 1) lsum += __shfl_xor(lsum, o, 32);
    if ((tid & 31) == 0) atomicAdd(gsum, lsum);
    if (tid < TILE_N) {
      unsigned bi = (unsigned)(sbest[tid] & 0xFFFFFFFFull);
      idx_out[n0 + tid] = (float)bi;
      atomicAdd(&counts[bi], 1.0f);
    }
  }
}

// ---------------- finalize: vq_loss + perplexity ----------------------------
__global__ void vq_fin(const float* __restrict__ counts,
                       const float* __restrict__ gsum,
                       float* __restrict__ out_scalars) {
  const int t = threadIdx.x;
  float ent = 0.f;
  for (int k = t; k < NUM_E; k += 256) {
    float p = counts[k] * (1.0f / (float)N_ROWS);
    ent += p * logf(p + 1e-10f);
  }
  for (int o = 16; o; o >>= 1) ent += __shfl_xor(ent, o, 32);
  __shared__ float ws[8];
  if ((t & 31) == 0) ws[t >> 5] = ent;
  __syncthreads();
  if (t == 0) {
    float s = 0.f;
    #pragma unroll
    for (int i = 0; i < 8; ++i) s += ws[i];
    out_scalars[0] = 1.25f * (*gsum) / (float)ZQ_ELEMS;  // (1+0.25)*MSE
    out_scalars[1] = expf(-s);
  }
}

extern "C" void kernel_launch(void* const* d_in, const int* in_sizes, int n_in,
                              void* d_out, int out_size, void* d_ws,
                              size_t ws_size, hipStream_t stream) {
  const float* z   = (const float*)d_in[0];
  const float* emb = (const float*)d_in[1];
  float* out = (float*)d_out;
  char* ws = (char*)d_ws;
  unsigned short* ehi = (unsigned short*)(ws);
  unsigned short* elo = (unsigned short*)(ws + 524288);
  float* esq    = (float*)(ws + 1048576);
  float* counts = (float*)(ws + 1052672);
  float* gsum   = (float*)(ws + 1056768);

  vq_prep<<<NUM_E, 256, 0, stream>>>(emb, ehi, elo, esq, counts, gsum);
  vq_main<<<N_ROWS / TILE_N, 256, 0, stream>>>(
      z, emb, ehi, elo, esq, out, out + ZQ_ELEMS + 2, counts, gsum);
  vq_fin<<<1, 256, 0, stream>>>(counts, gsum, out + ZQ_ELEMS);
}
